// BasicRNN_19310172963274
// MI455X (gfx1250) — matrix operand order repaired and checked by reading, not compiled, not run on hardware
//
#include <hip/hip_runtime.h>
#include <hip/hip_bf16.h>
#include <math.h>

#define BATCH 15
#define SEQ   4096
#define N_IN  512
#define N_H   1024
#define NWG   16          // persistent workgroups; each owns 64 output columns
#define NCOLS 64          // N_H / NWG
#define TPB   256         // 8 waves: 4 n-tiles x 2 K-halves

typedef __attribute__((ext_vector_type(16))) __bf16 v16bf;
typedef __attribute__((ext_vector_type(8)))  float  v8f;

// ---------------------------------------------------------------------------
// Fragment loaders from LDS, matching CDNA5 WMMA VGPR layouts (ISA 7.12.2).
// A (16x32 bf16, row-major LDS [16][ld]):
//   lanes 0-15 : row=lane,     VGPR0-3 = K 0..7,  VGPR4-7 = K 16..23
//   lanes16-31 : row=lane-16,  VGPR0-3 = K 8..15, VGPR4-7 = K 24..31
// ---------------------------------------------------------------------------
__device__ __forceinline__ v16bf load_a_frag(const __bf16* base, int ld, int kb, int lane) {
    int m    = lane & 15;
    int koff = (lane & 16) ? 8 : 0;
    const __bf16* p = base + m * ld + kb + koff;
    union { v16bf v; uint4 q[2]; } u;
    u.q[0] = *(const uint4*)(p);        // 8 bf16 = 16B -> ds_load_b128
    u.q[1] = *(const uint4*)(p + 16);
    return u.v;
}

// B (32x16 bf16). LDS holds the weight slice COLUMN-major: w[c][k], ldk = K.
//   lanes 0-15 : col=lane,     K 0..7,  K 16..23  -> two b128 loads
//   lanes16-31 : col=lane-16,  K 8..15, K 24..31
__device__ __forceinline__ v16bf load_b_frag(const __bf16* base, int ldk, int ncol_base,
                                             int kb, int lane) {
    int n    = lane & 15;
    int koff = (lane & 16) ? 8 : 0;
    const __bf16* p = base + (size_t)(ncol_base + n) * ldk + kb + koff;
    union { v16bf v; uint4 q[2]; } u;
    u.q[0] = *(const uint4*)(p);
    u.q[1] = *(const uint4*)(p + 16);
    return u.v;
}

__device__ __forceinline__ v8f wmma_bf16(v16bf a, v16bf b, v8f c) {
    return __builtin_amdgcn_wmma_f32_16x16x32_bf16(false, a, false, b, (short)0, c,
                                                   false, false);
}

// Monotonic grid barrier; counter is zeroed by a prologue kernel each launch,
// so target = (old/NWG + 1) * NWG is exact and graph replays stay deterministic.
__device__ __forceinline__ void grid_barrier(unsigned long long* ctr) {
    __syncthreads();
    if (threadIdx.x == 0) {
        __threadfence();                                  // release h stores
        unsigned long long old = atomicAdd(ctr, 1ull);
        unsigned long long tgt = (old / NWG + 1ull) * NWG;
        while (__hip_atomic_load(ctr, __ATOMIC_RELAXED, __HIP_MEMORY_SCOPE_AGENT) < tgt) {
            __builtin_amdgcn_s_sleep(2);
        }
        __threadfence();                                  // acquire (global_inv)
    }
    __syncthreads();
}

__global__ void rnn_barrier_init(unsigned long long* ctr) { *ctr = 0ull; }

// ---------------------------------------------------------------------------
// Persistent scan kernel: 16 WGs, weights resident in LDS, h exchanged via L2.
// ---------------------------------------------------------------------------
__global__ __launch_bounds__(TPB)
void rnn_scan_kernel(const float* __restrict__ x,
                     const float* __restrict__ W_ih,
                     const float* __restrict__ b_ih,
                     const float* __restrict__ W_hh,
                     const float* __restrict__ b_hh,
                     __bf16* __restrict__ hbuf0,
                     __bf16* __restrict__ hbuf1,
                     unsigned long long* __restrict__ ctr) {
    extern __shared__ __align__(16) char smem[];
    __bf16* whh_lds  = (__bf16*)(smem);                 // [64][1024]  131072 B
    __bf16* wih_lds  = (__bf16*)(smem + 131072);        // [64][512]    65536 B
    __bf16* h_lds    = (__bf16*)(smem + 196608);        // [16][1024]   32768 B
    __bf16* x_lds    = (__bf16*)(smem + 229376);        // [16][512]    16384 B
    float*  bias_lds = (float*) (smem + 245760);        // [64]           256 B
    float*  red      = (float*) (smem + 246016);        // [4][32][8]    4096 B

    const int tid  = threadIdx.x;
    const int lane = tid & 31;
    const int w    = tid >> 5;          // wave 0..7
    const int nt   = w & 3;             // n-tile within the 64 columns
    const int kh   = w >> 2;            // K half
    const int n0   = blockIdx.x * NCOLS;

    // ---- one-time weight staging: fp32 -> bf16, column-major in LDS --------
    for (int i = tid; i < NCOLS * N_H; i += TPB) {
        int c = i >> 10, k = i & 1023;
        whh_lds[i] = (__bf16)W_hh[(size_t)(n0 + c) * N_H + k];   // B[k][n]=W_hh[n][k]
    }
    for (int i = tid; i < NCOLS * N_IN; i += TPB) {
        int c = i >> 9, k = i & 511;
        wih_lds[i] = (__bf16)W_ih[(size_t)(n0 + c) * N_IN + k];
    }
    if (tid < NCOLS) bias_lds[tid] = b_ih[n0 + tid] + b_hh[n0 + tid];
    __syncthreads();

    __bf16* bufs[2] = { hbuf0, hbuf1 };
    int p = 0;                           // previous h lives in bufs[p]

    for (int t = 0; t < SEQ; ++t) {
        // ---- stage x_t (fp32->bf16, pad row 15 with zeros) -----------------
        for (int i = tid; i < 16 * N_IN; i += TPB) {
            int b = i >> 9, k = i & 511;
            float v = (b < BATCH) ? x[((size_t)b * SEQ + t) * N_IN + k] : 0.0f;
            x_lds[i] = (__bf16)v;
            if (t + 1 < SEQ && b < BATCH)            // warm L2 for next step
                __builtin_prefetch(&x[((size_t)b * SEQ + t + 1) * N_IN + k], 0, 1);
        }
        // ---- stage h_{t} from global (skip at t==0: h0 == 0) ---------------
        if (t > 0) {
            const uint* src = (const uint*)bufs[p];
            uint*       dst = (uint*)h_lds;
            for (int i = tid; i < 16 * N_H / 2; i += TPB) dst[i] = src[i];
        }
        __syncthreads();

        // ---- D = x_t @ W_ih_sliceT  (+ h @ W_hh_sliceT if t>0) -------------
        v8f acc0 = {}, acc1 = {};
        const int nb = nt * 16;
        for (int i = 0; i < 8; i += 2) {             // K half of 256 over x
            int kb = kh * 256 + i * 32;
            acc0 = wmma_bf16(load_a_frag(x_lds, N_IN, kb,      lane),
                             load_b_frag(wih_lds, N_IN, nb, kb,      lane), acc0);
            acc1 = wmma_bf16(load_a_frag(x_lds, N_IN, kb + 32, lane),
                             load_b_frag(wih_lds, N_IN, nb, kb + 32, lane), acc1);
        }
        if (t > 0) {
            for (int i = 0; i < 16; i += 2) {        // K half of 512 over h
                int kb = kh * 512 + i * 32;
                acc0 = wmma_bf16(load_a_frag(h_lds, N_H, kb,      lane),
                                 load_b_frag(whh_lds, N_H, nb, kb,      lane), acc0);
                acc1 = wmma_bf16(load_a_frag(h_lds, N_H, kb + 32, lane),
                                 load_b_frag(whh_lds, N_H, nb, kb + 32, lane), acc1);
            }
        }
        v8f acc = acc0 + acc1;

        // ---- cross-K-half reduction, bias, tanh, publish h_{t+1} -----------
        if (kh == 1) {
            float* r = &red[((nt * 32) + lane) * 8];
            #pragma unroll
            for (int i = 0; i < 8; ++i) r[i] = acc[i];
        }
        __syncthreads();
        if (kh == 0) {
            const float* r = &red[((nt * 32) + lane) * 8];
            int   col  = nb + (lane & 15);
            float bias = bias_lds[col];
            __bf16* hdst = bufs[p ^ 1];
            int mbase = (lane & 16) ? 8 : 0;         // C layout: lanes16-31 -> M+8
            #pragma unroll
            for (int i = 0; i < 8; ++i) {
                float v = tanhf(acc[i] + r[i] + bias);
                hdst[(size_t)(mbase + i) * N_H + (n0 + col)] = (__bf16)v;
            }
        }
        grid_barrier(ctr);
        p ^= 1;
    }
    // SEQ is even -> final h ends up in hbuf0 (p returns to 0).
}

// ---------------------------------------------------------------------------
// out[b] = sigmoid(h_final[b,:] . W_fc + b_fc)
// ---------------------------------------------------------------------------
__global__ void rnn_fc_kernel(const __bf16* __restrict__ h,
                              const float* __restrict__ W_fc,
                              const float* __restrict__ b_fc,
                              float* __restrict__ out) {
    int b = blockIdx.x;
    int tid = threadIdx.x;                // 128 threads
    float s = 0.0f;
    for (int k = tid; k < N_H; k += 128) s += (float)h[(size_t)b * N_H + k] * W_fc[k];
    #pragma unroll
    for (int off = 16; off > 0; off >>= 1) s += __shfl_down(s, off, 32);
    __shared__ float part[4];
    if ((tid & 31) == 0) part[tid >> 5] = s;
    __syncthreads();
    if (tid == 0) {
        float acc = part[0] + part[1] + part[2] + part[3] + b_fc[0];
        out[b] = 1.0f / (1.0f + expf(-acc));
    }
}

extern "C" void kernel_launch(void* const* d_in, const int* in_sizes, int n_in,
                              void* d_out, int out_size, void* d_ws, size_t ws_size,
                              hipStream_t stream) {
    const float* x    = (const float*)d_in[0];
    const float* W_ih = (const float*)d_in[1];
    const float* b_ih = (const float*)d_in[2];
    const float* W_hh = (const float*)d_in[3];
    const float* b_hh = (const float*)d_in[4];
    const float* W_fc = (const float*)d_in[5];
    const float* b_fc = (const float*)d_in[6];
    float* out = (float*)d_out;

    char* ws = (char*)d_ws;
    unsigned long long* ctr = (unsigned long long*)ws;          // 8 B
    __bf16* hbuf0 = (__bf16*)(ws + 64);                         // 16*1024*2 B
    __bf16* hbuf1 = (__bf16*)(ws + 64 + 16 * N_H * 2);          // 16*1024*2 B

    rnn_barrier_init<<<1, 1, 0, stream>>>(ctr);

    constexpr size_t lds_bytes = 131072 + 65536 + 32768 + 16384 + 256 + 4096;
    rnn_scan_kernel<<<NWG, TPB, lds_bytes, stream>>>(x, W_ih, b_ih, W_hh, b_hh,
                                                     hbuf0, hbuf1, ctr);

    rnn_fc_kernel<<<BATCH, 128, 0, stream>>>(hbuf0, W_fc, b_fc, out);
}
